// SpikeGraphTransformer_26465588478219
// MI455X (gfx1250) — matrix operand order repaired and checked by reading, not compiled, not run on hardware
//
#include <hip/hip_runtime.h>

// ---------------------------------------------------------------------------
// SpikeGraphTransformer for MI455X (gfx1250).
// Dense layers: bf16 WMMA (v_wmma_f32_16x16x32_bf16), fp32 accum, LDS tiles
// double-buffered and filled with GLOBAL_LOAD_ASYNC_TO_LDS_B128 (ASYNCcnt).
// BN/LIF/scatter are bandwidth-bound elementwise kernels. GEMM biases feeding
// BatchNorm (gamma=1, beta=0) cancel exactly and are dropped; only head_b is
// applied.
// ---------------------------------------------------------------------------

#define N_NODES 50000
#define N_EDGES 800000
#define F_INF   128
#define HIDD    64
#define HEADSN  8
#define DQKV    512   // HID*HEADS
#define OUT_DIM 47
#define BN_EPS  1e-5f

#define USE_ASYNC_LDS 1   // gfx1250 async global->LDS copies (inline asm)

typedef __bf16 bf16;
typedef __attribute__((ext_vector_type(16))) __bf16 bf16x16;
typedef __attribute__((ext_vector_type(8)))  float  f32x8;

// ---------------------------------------------------------------------------
// WMMA GEMM: C[M,N] = A[M,K] * W[N,K]^T (+bias). A,W bf16 row-major, C fp32.
// Requires N%64==0, K%32==0; M is guarded at the store (A-tile tail rows are
// clamped to row M-1, which is safe because C row r depends only on A row r).
// Block = 256 threads = 8 waves; block tile 128x64; each wave computes a
// 32x32 patch as 2x2 WMMA 16x16x32 tiles; K staged in double-buffered LDS.
// ---------------------------------------------------------------------------
__global__ __launch_bounds__(256) void k_gemm_bf16_wmma(
    const bf16* __restrict__ A, const bf16* __restrict__ W,
    const float* __restrict__ bias, float* __restrict__ C,
    int M, int N, int K)
{
  __shared__ __align__(16) bf16 sA[2][128][32];
  __shared__ __align__(16) bf16 sB[2][64][32];
  const int tid  = threadIdx.x;
  const int row0 = blockIdx.y * 128;
  const int col0 = blockIdx.x * 64;
  const int wave = tid >> 5, lane = tid & 31;
  const int wm = wave >> 1, wn = wave & 1;   // 4 x 2 wave grid
  const int l15 = lane & 15, kh = lane >> 4;

  f32x8 acc[2][2] = {};

  const int ar = tid >> 1, ah = tid & 1;     // A-tile load coords (row, 16-half)
  const int bn_ = tid >> 2, bq = tid & 3;    // B-tile load coords (row, 8-chunk)

  // Clamp tail rows: rows >= M read row M-1; their C rows are never stored.
  const long arow = ((long)row0 + ar < M) ? ((long)row0 + ar) : (long)(M - 1);
  const bf16* aptr = A + arow * K + ah * 16;            // 32B per thread
  const bf16* bptr = W + (long)(col0 + bn_) * K + bq * 8; // 16B per thread

#if USE_ASYNC_LDS
  // Raw LDS byte offsets (generic LDS pointer low 32 bits == LDS address).
  const unsigned ldsA[2] = {
      (unsigned)(unsigned long long)&sA[0][ar][ah * 16],
      (unsigned)(unsigned long long)&sA[1][ar][ah * 16]};
  const unsigned ldsB[2] = {
      (unsigned)(unsigned long long)&sB[0][bn_][bq * 8],
      (unsigned)(unsigned long long)&sB[1][bn_][bq * 8]};

  auto issue_fill = [&](int buf, int k0) {
    unsigned long long ga = (unsigned long long)(aptr + k0);
    unsigned long long gb = (unsigned long long)(bptr + k0);
    // INST_OFFSET applies to both the global and the LDS address (ISA 10.x),
    // so one base covers both 16B halves of the A chunk.
    asm volatile("global_load_async_to_lds_b128 %0, %1, off"
                 :: "v"(ldsA[buf]), "v"(ga) : "memory");
    asm volatile("global_load_async_to_lds_b128 %0, %1, off offset:16"
                 :: "v"(ldsA[buf]), "v"(ga) : "memory");
    asm volatile("global_load_async_to_lds_b128 %0, %1, off"
                 :: "v"(ldsB[buf]), "v"(gb) : "memory");
  };

  issue_fill(0, 0);
  int buf = 0;
  for (int k0 = 0; k0 < K; k0 += 32) {
    if (k0 + 32 < K) {
      issue_fill(buf ^ 1, k0 + 32);
      // 3 newly issued ops outstanding; async loads complete in order, so
      // asynccnt<=3 guarantees the current buffer is fully resident.
      asm volatile("s_wait_asynccnt 0x3" ::: "memory");
    } else {
      asm volatile("s_wait_asynccnt 0x0" ::: "memory");
    }
    __syncthreads();

    union Frag { bf16x16 v; float4 f4[2]; };
    Frag fa[2], fb[2];
#pragma unroll
    for (int i = 0; i < 2; ++i) {
      int m = 32 * wm + 16 * i + l15;
      fa[i].f4[0] = *reinterpret_cast<const float4*>(&sA[buf][m][kh * 8]);
      fa[i].f4[1] = *reinterpret_cast<const float4*>(&sA[buf][m][16 + kh * 8]);
      int n = 32 * wn + 16 * i + l15;
      fb[i].f4[0] = *reinterpret_cast<const float4*>(&sB[buf][n][kh * 8]);
      fb[i].f4[1] = *reinterpret_cast<const float4*>(&sB[buf][n][16 + kh * 8]);
    }
#pragma unroll
    for (int i = 0; i < 2; ++i)
#pragma unroll
      for (int j = 0; j < 2; ++j)
        acc[i][j] = __builtin_amdgcn_wmma_f32_16x16x32_bf16(
            false, fa[i].v, false, fb[j].v, (short)0, acc[i][j], false, false);
    __syncthreads();  // all waves done reading buf before it is refilled
    buf ^= 1;
  }
#else
  for (int k0 = 0; k0 < K; k0 += 32) {
    {
      const float4* p = reinterpret_cast<const float4*>(aptr + k0);
      *reinterpret_cast<float4*>(&sA[0][ar][ah * 16])     = p[0];
      *reinterpret_cast<float4*>(&sA[0][ar][ah * 16 + 8]) = p[1];
      if (k0 + 32 < K) __builtin_prefetch(aptr + k0 + 32, 0, 1);
    }
    {
      const float4* p = reinterpret_cast<const float4*>(bptr + k0);
      *reinterpret_cast<float4*>(&sB[0][bn_][bq * 8]) = p[0];
    }
    __syncthreads();
    union Frag { bf16x16 v; float4 f4[2]; };
    Frag fa[2], fb[2];
#pragma unroll
    for (int i = 0; i < 2; ++i) {
      int m = 32 * wm + 16 * i + l15;
      fa[i].f4[0] = *reinterpret_cast<const float4*>(&sA[0][m][kh * 8]);
      fa[i].f4[1] = *reinterpret_cast<const float4*>(&sA[0][m][16 + kh * 8]);
      int n = 32 * wn + 16 * i + l15;
      fb[i].f4[0] = *reinterpret_cast<const float4*>(&sB[0][n][kh * 8]);
      fb[i].f4[1] = *reinterpret_cast<const float4*>(&sB[0][n][16 + kh * 8]);
    }
#pragma unroll
    for (int i = 0; i < 2; ++i)
#pragma unroll
      for (int j = 0; j < 2; ++j)
        acc[i][j] = __builtin_amdgcn_wmma_f32_16x16x32_bf16(
            false, fa[i].v, false, fb[j].v, (short)0, acc[i][j], false, false);
    __syncthreads();
  }
#endif

  // Store C (16x16 f32 layout: vgpr r -> row (kh?8:0)+r, lane&15 -> col).
#pragma unroll
  for (int i = 0; i < 2; ++i)
#pragma unroll
    for (int j = 0; j < 2; ++j) {
      int nn = col0 + 32 * wn + 16 * j + l15;
      float b = bias ? bias[nn] : 0.0f;
      int mbase = row0 + 32 * wm + 16 * i + (kh ? 8 : 0);
#pragma unroll
      for (int r = 0; r < 8; ++r) {
        int m = mbase + r;
        if (m < M) C[(long)m * N + nn] = acc[i][j][r] + b;
      }
    }
}

// ---------------------------------------------------------------------------
// Elementwise / reduction kernels
// ---------------------------------------------------------------------------
__global__ void k_cvt_bf16(const float* __restrict__ in, bf16* __restrict__ out,
                           long n) {
  for (long i = blockIdx.x * 256L + threadIdx.x; i < n; i += 256L * gridDim.x)
    out[i] = (bf16)in[i];
}

// proj_W is [L,64,512] with col = h*64+d; permute to col' = d*8+h so that it
// matches the activation layout we keep (y columns are d*8+h).
__global__ void k_cvt_proj(const float* __restrict__ w, bf16* __restrict__ o,
                           long total) {
  for (long i = blockIdx.x * 256L + threadIdx.x; i < total; i += 256L * gridDim.x) {
    int c  = (int)(i & 511);          // permuted column index c' = d*8+h
    long nl = i >> 9;                 // l*64 + n
    int d = c >> 3, hh = c & 7;
    o[i] = (bf16)w[nl * 512 + (long)hh * 64 + d];
  }
}

__global__ void k_copy_f32(const float* __restrict__ s, float* __restrict__ d,
                           long n) {
  for (long i = blockIdx.x * 256L + threadIdx.x; i < n; i += 256L * gridDim.x)
    d[i] = s[i];
}

// dst[0..n) = src, dst[n..2n) = src  (broadcast over T=2)
__global__ void k_bcast2(const float* __restrict__ s, float* __restrict__ d,
                         long n) {
  for (long i = blockIdx.x * 256L + threadIdx.x; i < n; i += 256L * gridDim.x) {
    float v = s[i];
    d[i] = v; d[i + n] = v;
  }
}

// Column mean / rstd over Mrows (training-mode BN, biased variance).
__global__ __launch_bounds__(256) void k_bn_stats(
    const float* __restrict__ X, float* __restrict__ mean,
    float* __restrict__ rstd, int Mrows, int Ncols) {
  int col = blockIdx.x;
  float s = 0.f, ss = 0.f;
  for (int r = threadIdx.x; r < Mrows; r += 256) {
    float x = X[(long)r * Ncols + col];
    s += x; ss += x * x;
  }
  __shared__ float shs[256], shq[256];
  shs[threadIdx.x] = s; shq[threadIdx.x] = ss;
  __syncthreads();
  for (int off = 128; off > 0; off >>= 1) {
    if (threadIdx.x < off) {
      shs[threadIdx.x] += shs[threadIdx.x + off];
      shq[threadIdx.x] += shq[threadIdx.x + off];
    }
    __syncthreads();
  }
  if (threadIdx.x == 0) {
    float m = shs[0] / (float)Mrows;
    float v = shq[0] / (float)Mrows - m * m;
    mean[col] = m;
    rstd[col] = rsqrtf(fmaxf(v, 0.f) + BN_EPS);
  }
}

// y = (x-mean)*rstd ; optional relu ; optional += residual ; fp32 out.
// (no __restrict__: callers use this in place, X==Y / residual==Y)
__global__ void k_bn_apply(const float* X, const float* __restrict__ mean,
                           const float* __restrict__ rstd,
                           const float* residual, float* Y, long Mrows,
                           int Ncols, int relu) {
  long total = Mrows * (long)Ncols;
  for (long i = blockIdx.x * 256L + threadIdx.x; i < total; i += 256L * gridDim.x) {
    int col = (int)(i % Ncols);
    float y = (X[i] - mean[col]) * rstd[col];
    if (relu) y = fmaxf(y, 0.f);
    if (residual) y += residual[i];
    Y[i] = y;
  }
}

// Closed-form 2-step LIF (tau=2, hard reset): x laid out [2, Ntot].
// (no __restrict__ on x/sf: the kv call runs in place)
__global__ void k_lif2(const float* x, float* sf, bf16* __restrict__ sb,
                       long Ntot, float th) {
  for (long i = blockIdx.x * 256L + threadIdx.x; i < Ntot; i += 256L * gridDim.x) {
    float x0 = x[i], x1 = x[i + Ntot];
    float v = 0.5f * x0;
    float s0 = (v >= th) ? 1.f : 0.f;
    v = v * (1.f - s0);
    v = 0.5f * (v + x1);
    float s1 = (v >= th) ? 1.f : 0.f;
    if (sf) { sf[i] = s0; sf[i + Ntot] = s1; }
    if (sb) { sb[i] = (bf16)s0; sb[i + Ntot] = (bf16)s1; }
  }
}

__global__ void k_deg(const int* __restrict__ col, float* __restrict__ deg,
                      long e) {
  for (long i = blockIdx.x * 256L + threadIdx.x; i < e; i += 256L * gridDim.x)
    atomicAdd(&deg[col[i]], 1.f);
}

__global__ void k_dinv(float* __restrict__ deg, long n) {
  for (long i = blockIdx.x * 256L + threadIdx.x; i < n; i += 256L * gridDim.x) {
    float d = deg[i];
    deg[i] = (d > 0.f) ? rsqrtf(d) : 0.f;
  }
}

// agg[col] += dinv[col]*dinv[row]*h[row], 64 features/edge.
__global__ void k_edge_agg(const int* __restrict__ row,
                           const int* __restrict__ col,
                           const float* __restrict__ dinv,
                           const float* __restrict__ h,
                           float* __restrict__ agg, long total) {
  for (long i = blockIdx.x * 256L + threadIdx.x; i < total; i += 256L * gridDim.x) {
    long e = i >> 6;
    int f = (int)(i & 63);
    int r = row[e], c = col[e];
    float val = dinv[c] * dinv[r];
    atomicAdd(&agg[(long)c * 64 + f], val * h[(long)r * 64 + f]);
  }
}

// kvsum[t,b,d] = sum_h k[t,b,d*8+h]*v[t,b,d*8+h]  (contiguous 8-wide)
__global__ void k_kvsum(const bf16* __restrict__ ks, const bf16* __restrict__ vs,
                        float* __restrict__ kv, long total) {
  for (long i = blockIdx.x * 256L + threadIdx.x; i < total; i += 256L * gridDim.x) {
    long base = i * 8;
    float s = 0.f;
#pragma unroll
    for (int h = 0; h < 8; ++h)
      s += (float)ks[base + h] * (float)vs[base + h];
    kv[i] = s;
  }
}

// a = q * kv (broadcast over heads), in place on the q-spike buffer.
// flat index i = (t*N+b)*512 + d*8 + h ; kv index = i >> 3.
__global__ void k_amul(bf16* __restrict__ qs, const float* __restrict__ kv,
                       long total) {
  for (long i = blockIdx.x * 256L + threadIdx.x; i < total; i += 256L * gridDim.x)
    qs[i] = (bf16)((float)qs[i] * kv[i >> 3]);
}

// out[n,o] = bias[o] + sum_k 0.5*(xt0[n,k]+xt1[n,k]) * headW[o,k]
__global__ void k_head(const float* __restrict__ xt, const float* __restrict__ Wh,
                       const float* __restrict__ bh, float* __restrict__ out,
                       long Nn) {
  long total = Nn * OUT_DIM;
  for (long i = blockIdx.x * 256L + threadIdx.x; i < total; i += 256L * gridDim.x) {
    long n = i / OUT_DIM;
    int o = (int)(i % OUT_DIM);
    const float* x0 = xt + n * HIDD;
    const float* x1 = xt + Nn * HIDD + n * HIDD;
    const float* w = Wh + (long)o * HIDD;
    float s = bh[o];
#pragma unroll 8
    for (int k = 0; k < HIDD; ++k) s += 0.5f * (x0[k] + x1[k]) * w[k];
    out[i] = s;
  }
}

// ---------------------------------------------------------------------------
// Host-side orchestration
// ---------------------------------------------------------------------------
static inline int ew_grid(long n) {
  long b = (n + 255) / 256;
  return (int)(b < 16384 ? b : 16384);
}

extern "C" void kernel_launch(void* const* d_in, const int* in_sizes, int n_in,
                              void* d_out, int out_size, void* d_ws, size_t ws_size,
                              hipStream_t stream) {
  (void)in_sizes; (void)n_in; (void)out_size; (void)ws_size;

  const float* x      = (const float*)d_in[0];
  const int*   ei     = (const int*)d_in[1];   // [2,E]: rows then cols
  const float* lin_W  = (const float*)d_in[2];
  const float* conv_W = (const float*)d_in[4];
  const float* qkv_W  = (const float*)d_in[6];
  const float* proj_W = (const float*)d_in[8];
  const float* fc1_W  = (const float*)d_in[10];
  const float* fc2_W  = (const float*)d_in[12];
  const float* head_W = (const float*)d_in[14];
  const float* head_b = (const float*)d_in[15];
  float* out = (float*)d_out;

  const long Nn = N_NODES, Ee = N_EDGES, TN = 2L * N_NODES;
  const int* e_row = ei;
  const int* e_col = ei + Ee;

  // ---- workspace carve (256B aligned) ----
  size_t off = 0;
  auto carve = [&](size_t bytes) -> char* {
    char* p = (char*)d_ws + off;
    off += (bytes + 255) & ~(size_t)255;
    return p;
  };
  bf16* w_lin  = (bf16*)carve(64L * 128 * 2);
  bf16* w_conv = (bf16*)carve(2L * 64 * 64 * 2);
  bf16* w_qkv  = (bf16*)carve(2L * 3 * 512 * 64 * 2);
  bf16* w_proj = (bf16*)carve(2L * 64 * 512 * 2);
  bf16* w_fc1  = (bf16*)carve(2L * 128 * 64 * 2);
  bf16* w_fc2  = (bf16*)carve(2L * 64 * 128 * 2);
  bf16* xb     = (bf16*)carve(Nn * F_INF * 2);
  float* h     = (float*)carve(Nn * HIDD * 4);
  float* h0    = (float*)carve(Nn * HIDD * 4);
  float* agg   = (float*)carve(Nn * HIDD * 4);
  bf16*  aggb  = (bf16*)carve(Nn * HIDD * 2);
  float* dinv  = (float*)carve(Nn * 4);
  float* xt    = (float*)carve(TN * HIDD * 4);
  bf16*  sb    = (bf16*)carve(TN * 128 * 2);       // spike input, widest = 128
  float* y     = (float*)carve(TN * DQKV * 4);     // GEMM/BN scratch, widest = 512
  bf16*  qs    = (bf16*)carve(TN * DQKV * 2);
  bf16*  ks    = (bf16*)carve(TN * DQKV * 2);
  bf16*  vs    = (bf16*)carve(TN * DQKV * 2);
  float* kv    = (float*)carve(TN * HIDD * 4);
  float* mean  = (float*)carve(512 * 4);
  float* rstd  = (float*)carve(512 * 4);

  auto gemm = [&](const bf16* A, const bf16* W, const float* bias, float* C,
                  int M, int N, int K) {
    dim3 g(N / 64, (M + 127) / 128);
    k_gemm_bf16_wmma<<<g, 256, 0, stream>>>(A, W, bias, C, M, N, K);
  };
  auto bn = [&](const float* X, const float* res, float* Y, long M, int N,
                int relu) {
    k_bn_stats<<<N, 256, 0, stream>>>(X, mean, rstd, (int)M, N);
    k_bn_apply<<<ew_grid(M * N), 256, 0, stream>>>(X, mean, rstd, res, Y, M, N,
                                                   relu);
  };

  // ---- weight & input conversion ----
  k_cvt_bf16<<<ew_grid(Nn * F_INF), 256, 0, stream>>>(x, xb, Nn * F_INF);
  k_cvt_bf16<<<ew_grid(64L * 128), 256, 0, stream>>>(lin_W, w_lin, 64L * 128);
  k_cvt_bf16<<<ew_grid(2L * 64 * 64), 256, 0, stream>>>(conv_W, w_conv, 2L * 64 * 64);
  k_cvt_bf16<<<ew_grid(2L * 3 * 512 * 64), 256, 0, stream>>>(qkv_W, w_qkv, 2L * 3 * 512 * 64);
  k_cvt_proj<<<ew_grid(2L * 64 * 512), 256, 0, stream>>>(proj_W, w_proj, 2L * 64 * 512);
  k_cvt_bf16<<<ew_grid(2L * 128 * 64), 256, 0, stream>>>(fc1_W, w_fc1, 2L * 128 * 64);
  k_cvt_bf16<<<ew_grid(2L * 64 * 128), 256, 0, stream>>>(fc2_W, w_fc2, 2L * 64 * 128);

  // ---- GConv input projection: h0 = relu(bn(x @ linW^T)) (bias cancels in BN)
  gemm(xb, w_lin, nullptr, y, (int)Nn, HIDD, F_INF);
  bn(y, nullptr, h, Nn, HIDD, /*relu=*/1);
  k_copy_f32<<<ew_grid(Nn * HIDD), 256, 0, stream>>>(h, h0, Nn * HIDD);

  // ---- symmetric-normalized degrees
  hipMemsetAsync(dinv, 0, Nn * 4, stream);
  k_deg<<<ew_grid(Ee), 256, 0, stream>>>(e_col, dinv, Ee);
  k_dinv<<<ew_grid(Nn), 256, 0, stream>>>(dinv, Nn);

  // ---- 2 GNN layers: h = relu(bn(agg @ convW^T)) + h0
  for (int i = 0; i < 2; ++i) {
    hipMemsetAsync(agg, 0, Nn * HIDD * 4, stream);
    k_edge_agg<<<ew_grid(Ee * 64), 256, 0, stream>>>(e_row, e_col, dinv, h, agg,
                                                     Ee * 64);
    k_cvt_bf16<<<ew_grid(Nn * HIDD), 256, 0, stream>>>(agg, aggb, Nn * HIDD);
    gemm(aggb, w_conv + (long)i * 64 * 64, nullptr, y, (int)Nn, HIDD, HIDD);
    bn(y, h0, h, Nn, HIDD, /*relu=*/1);
  }

  // ---- xt[0] = xt[1] = h
  k_bcast2<<<ew_grid(Nn * HIDD), 256, 0, stream>>>(h, xt, Nn * HIDD);

  // ---- 2 spiking transformer blocks
  for (int l = 0; l < 2; ++l) {
    // MS_SSA_Conv
    k_lif2<<<ew_grid(Nn * HIDD), 256, 0, stream>>>(xt, nullptr, sb, Nn * HIDD, 1.0f);
    for (int j = 0; j < 3; ++j) {
      gemm(sb, w_qkv + ((long)l * 3 + j) * 512 * 64, nullptr, y, (int)TN, DQKV, HIDD);
      bn(y, nullptr, y, TN, DQKV, 0);
      bf16* dst = (j == 0) ? qs : (j == 1) ? ks : vs;
      k_lif2<<<ew_grid(Nn * DQKV), 256, 0, stream>>>(y, nullptr, dst, Nn * DQKV, 1.0f);
    }
    k_kvsum<<<ew_grid(TN * HIDD), 256, 0, stream>>>(ks, vs, kv, TN * HIDD);
    k_lif2<<<ew_grid(Nn * HIDD), 256, 0, stream>>>(kv, kv, nullptr, Nn * HIDD, 0.5f);
    k_amul<<<ew_grid(TN * DQKV), 256, 0, stream>>>(qs, kv, TN * DQKV);
    gemm(qs, w_proj + (long)l * 64 * 512, nullptr, y, (int)TN, HIDD, DQKV);
    bn(y, xt, xt, TN, HIDD, 0);  // + identity residual

    // MS_MLP_Conv
    k_lif2<<<ew_grid(Nn * HIDD), 256, 0, stream>>>(xt, nullptr, sb, Nn * HIDD, 1.0f);
    gemm(sb, w_fc1 + (long)l * 128 * 64, nullptr, y, (int)TN, 128, HIDD);
    bn(y, nullptr, y, TN, 128, 0);
    k_lif2<<<ew_grid(Nn * 128), 256, 0, stream>>>(y, nullptr, sb, Nn * 128, 1.0f);
    gemm(sb, w_fc2 + (long)l * 64 * 128, nullptr, y, (int)TN, HIDD, 128);
    bn(y, xt, xt, TN, HIDD, 0);  // + identity residual
  }

  // ---- head: mean over T then [N,64]@[64,47] + bias
  k_head<<<ew_grid(Nn * OUT_DIM), 256, 0, stream>>>(xt, head_W, head_b, out, Nn);
}
